// Mol_79319456023410
// MI455X (gfx1250) — matrix operand order repaired and checked by reading, not compiled, not run on hardware
//
#include <hip/hip_runtime.h>
#include <math.h>

// ---------------------------------------------------------------------------
// CDNA5 wave32 WMMA bf16:  D(16x16,f32) += A(16x32,bf16) x B(32x16,bf16)
// Operands are pre-packed into fragment order so the GEMM inner loop is
// branch-free: 4x global_load_b128 + 1x v_wmma per K-step.
// ---------------------------------------------------------------------------
typedef __attribute__((ext_vector_type(16))) __bf16 v16bf;
typedef __attribute__((ext_vector_type(8)))  float  v8f;

__device__ inline float lrelu_f(float x) { return x > 0.0f ? x : 0.01f * x; }

// A-fragment layout (ISA 7.12.2, 16-bit A 16x32): lane = hsel*16 + r,
//   element e:  k = kc*32 + ((e<8) ? hsel*8+e : 16+hsel*8+(e-8)),  row m = r
// Fragment buffer: [(mt*KC + kc)*32 + lane] -> v16bf
// B-fragment layout (32x16): lane holds column n = lane&15,
//   element e:  k = kc*32 + ((e<8) ? hsel*8+e : 16+hsel*8+(e-8))
// Fragment buffer: [(nt*KC + kc)*32 + lane] -> v16bf

// ---- pack weight W[Kact x NC] (f32 row-major) into B fragments ------------
__global__ void k_pack_w(const float* __restrict__ W, int Kact, int NC, int KC,
                         __bf16* __restrict__ outB) {
  int NT = NC >> 4;
  int tid = blockIdx.x * blockDim.x + threadIdx.x;
  int total = NT * KC * 32;
  if (tid >= total) return;
  int lane = tid & 31;
  int idx  = tid >> 5;
  int kc = idx % KC, nt = idx / KC;
  int n = (nt << 4) + (lane & 15);
  int hsel = lane >> 4;
  v16bf v;
#pragma unroll
  for (int e = 0; e < 16; ++e) {
    int k = kc * 32 + ((e < 8) ? hsel * 8 + e : 16 + hsel * 8 + (e - 8));
    float f = (k < Kact) ? W[(size_t)k * NC + n] : 0.0f;
    v[e] = (__bf16)f;
  }
  ((v16bf*)outB)[(size_t)idx * 32 + lane] = v;
}

// ---- pack activation A[M x ld] (f32 row-major) into A fragments -----------
__global__ void k_pack_a(const float* __restrict__ A, int ld, int M, int Kact,
                         int KC, __bf16* __restrict__ outA) {
  long long Mt = (M + 15) >> 4;
  long long tid = blockIdx.x * (long long)blockDim.x + threadIdx.x;
  long long total = Mt * KC * 32;
  if (tid >= total) return;
  int lane = (int)(tid & 31);
  long long idx = tid >> 5;
  int kc = (int)(idx % KC);
  long long mt = idx / KC;
  int r = lane & 15, hsel = lane >> 4;
  long long row = mt * 16 + r;
  if (row >= M) row = M - 1;  // clamp: rows >= M computed but never stored
  const float* Ar = A + (size_t)row * ld;
  int k0 = kc * 32 + hsel * 8;
  v16bf v;
#pragma unroll
  for (int j = 0; j < 8; ++j) {
    int ka = k0 + j;
    int kb = k0 + 16 + j;
    v[j]     = (__bf16)((ka < Kact) ? Ar[ka] : 0.0f);
    v[8 + j] = (__bf16)((kb < Kact) ? Ar[kb] : 0.0f);
  }
  ((v16bf*)outA)[idx * 32 + lane] = v;
}

// ---- pack edge-concat rows: row e = [node_feats[src[e]](74) | edge_feats[e](12)]
__global__ void k_pack_a_edge(const float* __restrict__ nf,
                              const float* __restrict__ ef,
                              const int* __restrict__ src, int E, int KC,
                              __bf16* __restrict__ outA) {
  long long Mt = (E + 15) >> 4;
  long long tid = blockIdx.x * (long long)blockDim.x + threadIdx.x;
  long long total = Mt * KC * 32;
  if (tid >= total) return;
  int lane = (int)(tid & 31);
  long long idx = tid >> 5;
  int kc = (int)(idx % KC);
  long long mt = idx / KC;
  int r = lane & 15, hsel = lane >> 4;
  long long e = mt * 16 + r;
  if (e >= E) e = E - 1;
  const float* nr = nf + (size_t)src[e] * 74;
  const float* er = ef + (size_t)e * 12;
  int k0 = kc * 32 + hsel * 8;
  v16bf v;
#pragma unroll
  for (int j = 0; j < 8; ++j) {
    int ka = k0 + j;
    int kb = k0 + 16 + j;
    float fa = (ka < 74) ? nr[ka] : ((ka < 86) ? er[ka - 74] : 0.0f);
    float fb = (kb < 74) ? nr[kb] : ((kb < 86) ? er[kb - 74] : 0.0f);
    v[j] = (__bf16)fa;
    v[8 + j] = (__bf16)fb;
  }
  ((v16bf*)outA)[idx * 32 + lane] = v;
}

// ---- packed GEMM: out[M x NC] = act(Afrag @ Bfrag + bias) ------------------
__global__ void k_gemm_packed(const __bf16* __restrict__ Afrag, int M,
                              const __bf16* __restrict__ Bfrag, int KC,
                              const float* __restrict__ bias, int NC,
                              float* __restrict__ out, int act) {
  int lane = threadIdx.x & 31;
  long long wave = blockIdx.x * (long long)(blockDim.x >> 5) + (threadIdx.x >> 5);
  int NT = NC >> 4;
  long long Mt = (M + 15) >> 4;
  if (wave >= Mt * NT) return;
  long long mt = wave / NT;
  int nt = (int)(wave % NT);
  const v16bf* Ap = (const v16bf*)Afrag + (mt * KC) * 32 + lane;
  const v16bf* Bp = (const v16bf*)Bfrag + ((long long)nt * KC) * 32 + lane;
  v8f acc = {};
#pragma unroll 4
  for (int kc = 0; kc < KC; ++kc) {
    v16bf a = Ap[kc * 32];
    v16bf b = Bp[kc * 32];
    acc = __builtin_amdgcn_wmma_f32_16x16x32_bf16(false, a, false, b, (short)0,
                                                  acc, false, false);
  }
  int r = lane & 15, hsel = lane >> 4;
  int n = (nt << 4) + r;
  float bn = bias[n];
#pragma unroll
  for (int i = 0; i < 8; ++i) {
    long long m = mt * 16 + i + 8 * hsel;
    if (m < M) {
      float v = acc[i] + bn;
      if (act == 1) v = lrelu_f(v);
      out[(size_t)m * NC + n] = v;
    }
  }
}

// ---- packed GEMM fused with attention scale + segment-sum scatter ----------
// ctx[dst[e]][n] += attn[e] * (he1 @ Wet + b)[e][n]
__global__ void k_gemm_msg_scatter(const __bf16* __restrict__ Afrag, int E,
                                   const __bf16* __restrict__ Bfrag, int KC,
                                   const float* __restrict__ bias,
                                   const float* __restrict__ attn,
                                   const int* __restrict__ dst,
                                   float* __restrict__ ctx) {
  int lane = threadIdx.x & 31;
  long long wave = blockIdx.x * (long long)(blockDim.x >> 5) + (threadIdx.x >> 5);
  int NT = 8;  // NC = 128
  long long Mt = (E + 15) >> 4;
  if (wave >= Mt * NT) return;
  long long mt = wave / NT;
  int nt = (int)(wave % NT);
  const v16bf* Ap = (const v16bf*)Afrag + (mt * KC) * 32 + lane;
  const v16bf* Bp = (const v16bf*)Bfrag + ((long long)nt * KC) * 32 + lane;
  v8f acc = {};
#pragma unroll 4
  for (int kc = 0; kc < KC; ++kc) {
    v16bf a = Ap[kc * 32];
    v16bf b = Bp[kc * 32];
    acc = __builtin_amdgcn_wmma_f32_16x16x32_bf16(false, a, false, b, (short)0,
                                                  acc, false, false);
  }
  int r = lane & 15, hsel = lane >> 4;
  int n = (nt << 4) + r;
  float bn = bias[n];
#pragma unroll
  for (int i = 0; i < 8; ++i) {
    long long m = mt * 16 + i + 8 * hsel;
    if (m < E) {
      float v = (acc[i] + bn) * attn[m];
      atomicAdd(&ctx[(size_t)dst[m] * 128 + n], v);
    }
  }
}

// ---------------------------------------------------------------------------
// Elementwise / softmax / scatter kernels
// ---------------------------------------------------------------------------
__device__ inline unsigned enc_f(float f) {
  unsigned u = __float_as_uint(f);
  return (u & 0x80000000u) ? ~u : (u | 0x80000000u);
}
__device__ inline float dec_f(unsigned u) {
  return (u & 0x80000000u) ? __uint_as_float(u & 0x7FFFFFFFu)
                           : __uint_as_float(~u);
}

__global__ void k_fill_f(float* p, float v, long long n) {
  long long i = blockIdx.x * (long long)blockDim.x + threadIdx.x;
  if (i < n) p[i] = v;
}
__global__ void k_fill_u(unsigned* p, unsigned v, long long n) {
  long long i = blockIdx.x * (long long)blockDim.x + threadIdx.x;
  if (i < n) p[i] = v;
}

// logits[e] = lrelu( dot(fd[dst[e]], W[0:128]) + dot(fs[idx], W[128:256]) + b )
__global__ void k_edge_logits(const float* __restrict__ fd,
                              const float* __restrict__ fs, int gather_src,
                              const int* __restrict__ src,
                              const int* __restrict__ dst,
                              const float* __restrict__ W,
                              const float* __restrict__ b,
                              float* __restrict__ logits, int E) {
  int e = blockIdx.x * blockDim.x + threadIdx.x;
  if (e >= E) return;
  const float4* pd = (const float4*)(fd + (size_t)dst[e] * 128);
  const float4* ps = (const float4*)(fs + (size_t)(gather_src ? src[e] : e) * 128);
  const float4* w0 = (const float4*)W;
  const float4* w1 = (const float4*)(W + 128);
  float acc = b[0];
#pragma unroll
  for (int c = 0; c < 32; ++c) {
    float4 d4 = pd[c], s4 = ps[c], a4 = w0[c], b4 = w1[c];
    acc += d4.x * a4.x + d4.y * a4.y + d4.z * a4.z + d4.w * a4.w;
    acc += s4.x * b4.x + s4.y * b4.y + s4.z * b4.z + s4.w * b4.w;
  }
  logits[e] = lrelu_f(acc);
}

__global__ void k_seg_max(const float* __restrict__ logits,
                          const int* __restrict__ dst, unsigned* menc, int E) {
  int e = blockIdx.x * blockDim.x + threadIdx.x;
  if (e >= E) return;
  atomicMax(&menc[dst[e]], enc_f(logits[e]));
}
__global__ void k_exp_sum(const float* __restrict__ logits,
                          const int* __restrict__ dst,
                          const unsigned* __restrict__ menc, float* attn,
                          float* ssum, int E) {
  int e = blockIdx.x * blockDim.x + threadIdx.x;
  if (e >= E) return;
  float m = dec_f(menc[dst[e]]);
  float ex = expf(logits[e] - m);
  attn[e] = ex;
  atomicAdd(&ssum[dst[e]], ex);
}
__global__ void k_norm(float* attn, const float* __restrict__ ssum,
                       const int* __restrict__ dst, int E) {
  int e = blockIdx.x * blockDim.x + threadIdx.x;
  if (e >= E) return;
  attn[e] /= ssum[dst[e]];
}

__global__ void k_elu(float* p, long long n) {
  long long i = blockIdx.x * (long long)blockDim.x + threadIdx.x;
  if (i >= n) return;
  float x = p[i];
  p[i] = x > 0.0f ? x : (expf(x) - 1.0f);
}

// ctx[dst[e]][c] += hp[src[e]][c] * attn[e]
__global__ void k_scatter_ctx(const float* __restrict__ hp,
                              const float* __restrict__ attn,
                              const int* __restrict__ src,
                              const int* __restrict__ dst, float* ctx,
                              long long total) {
  long long i = blockIdx.x * (long long)blockDim.x + threadIdx.x;
  if (i >= total) return;
  int e = (int)(i >> 7), c = (int)(i & 127);
  atomicAdd(&ctx[(size_t)dst[e] * 128 + c],
            hp[(size_t)src[e] * 128 + c] * attn[e]);
}

// h_out = relu(gru(gi, gh, h_prev)) elementwise over N*128
__global__ void k_gru(const float* __restrict__ gi, const float* __restrict__ gh,
                      const float* __restrict__ hprev, float* __restrict__ hout,
                      long long total) {
  long long i = blockIdx.x * (long long)blockDim.x + threadIdx.x;
  if (i >= total) return;
  int n = (int)(i >> 7), c = (int)(i & 127);
  size_t b3 = (size_t)n * 384;
  float ir = gi[b3 + c], iz = gi[b3 + 128 + c], inn = gi[b3 + 256 + c];
  float hr = gh[b3 + c], hz = gh[b3 + 128 + c], hn = gh[b3 + 256 + c];
  float r = 1.0f / (1.0f + expf(-(ir + hr)));
  float z = 1.0f / (1.0f + expf(-(iz + hz)));
  float nn = tanhf(inn + r * hn);
  float hp = hprev[i];
  float h = (1.0f - z) * nn + z * hp;
  hout[i] = h > 0.0f ? h : 0.0f;
}

__global__ void k_pool_add(const float* __restrict__ h,
                           const int* __restrict__ gid, float* out,
                           long long total) {
  long long i = blockIdx.x * (long long)blockDim.x + threadIdx.x;
  if (i >= total) return;
  int n = (int)(i >> 7), c = (int)(i & 127);
  atomicAdd(&out[(size_t)gid[n] * 128 + c], h[i]);
}
__global__ void k_counts(const int* __restrict__ gid, float* counts, int N) {
  int i = blockIdx.x * blockDim.x + threadIdx.x;
  if (i >= N) return;
  atomicAdd(&counts[gid[i]], 1.0f);
}
__global__ void k_finalize(float* out, const float* __restrict__ counts,
                           long long total) {
  long long i = blockIdx.x * (long long)blockDim.x + threadIdx.x;
  if (i >= total) return;
  out[i] /= fmaxf(counts[i >> 7], 1.0f);
}

// ---------------------------------------------------------------------------
// Host orchestration
// ---------------------------------------------------------------------------
static inline int gemm_blocks(long long M, int NC) {
  long long tiles = ((M + 15) / 16) * (NC / 16);
  return (int)((tiles + 7) / 8);  // 8 waves per 256-thread block
}
static inline int ew_blocks(long long n) { return (int)((n + 255) / 256); }
static inline long long afrag_floats(long long M, int KC) {
  return ((M + 15) / 16) * (long long)KC * 256;  // bf16 elems / 2
}
static inline long long wfrag_floats(int NC, int KC) {
  return (long long)(NC / 16) * KC * 256;
}

extern "C" void kernel_launch(void* const* d_in, const int* in_sizes, int n_in,
                              void* d_out, int out_size, void* d_ws,
                              size_t ws_size, hipStream_t stream) {
  const float* nf  = (const float*)d_in[0];
  const float* ef  = (const float*)d_in[1];
  const int*   src = (const int*)d_in[2];
  const int*   dst = (const int*)d_in[3];
  const int*   gid = (const int*)d_in[4];

  const int N = in_sizes[0] / 74;
  const int E = in_sizes[2];
  const int G = out_size / 128;

  const float* pnW   = (const float*)d_in[5];  const float* pnB   = (const float*)d_in[6];
  const float* pe1W  = (const float*)d_in[7];  const float* pe1B  = (const float*)d_in[8];
  const float* pe2W  = (const float*)d_in[9];  const float* pe2B  = (const float*)d_in[10];
  const float* etW   = (const float*)d_in[11]; const float* etB   = (const float*)d_in[12];
  const float* g0ihW = (const float*)d_in[13]; const float* g0ihB = (const float*)d_in[14];
  const float* g0hhW = (const float*)d_in[15]; const float* g0hhB = (const float*)d_in[16];

  // ---- workspace layout (float units) --
  float* ws = (float*)d_ws;
  size_t off = 0;
  float*    hv    = ws + off; off += (size_t)N * 128;
  float*    he1   = ws + off; off += (size_t)E * 128;
  float*    logit = ws + off; off += (size_t)E;
  float*    attn  = ws + off; off += (size_t)E;
  unsigned* menc  = (unsigned*)(ws + off); off += (size_t)N;
  float*    ssum  = ws + off; off += (size_t)N;
  float*    ctx   = ws + off; off += (size_t)N * 128;
  float*    gi    = ws + off; off += (size_t)N * 384;
  float*    gh    = ws + off; off += (size_t)N * 384;
  float*    h     = ws + off; off += (size_t)N * 128;
  float*    hp    = ws + off; off += (size_t)N * 128;
  float*    counts= ws + off; off += (size_t)G + 8;

  // packed activation fragments (bf16)
  __bf16* nfA  = (__bf16*)(ws + off); off += afrag_floats(N, 3);
  __bf16* xeA  = (__bf16*)(ws + off); off += afrag_floats(E, 3);
  __bf16* he1A = (__bf16*)(ws + off); off += afrag_floats(E, 4);
  __bf16* hvA  = (__bf16*)(ws + off); off += afrag_floats(N, 4);
  __bf16* ctxA = (__bf16*)(ws + off); off += afrag_floats(N, 4);
  __bf16* hA   = (__bf16*)(ws + off); off += afrag_floats(N, 4);
  // packed weight fragments (bf16)
  __bf16* pnWB    = (__bf16*)(ws + off); off += wfrag_floats(128, 3);
  __bf16* pe1WB   = (__bf16*)(ws + off); off += wfrag_floats(128, 3);
  __bf16* etWB    = (__bf16*)(ws + off); off += wfrag_floats(128, 4);
  __bf16* g0ihWB  = (__bf16*)(ws + off); off += wfrag_floats(384, 4);
  __bf16* g0hhWB  = (__bf16*)(ws + off); off += wfrag_floats(384, 4);
  __bf16* pnodeWB = (__bf16*)(ws + off); off += wfrag_floats(128, 4);
  __bf16* lihWB   = (__bf16*)(ws + off); off += wfrag_floats(384, 4);
  __bf16* lhhWB   = (__bf16*)(ws + off); off += wfrag_floats(384, 4);

  float* out = (float*)d_out;
  const long long NF = (long long)N * 128;
  const long long EF = (long long)E * 128;
  const dim3 B256(256);
  const unsigned NEG_INF_ENC = 0x007FFFFFu;  // enc(-inf)

  // ---- pack static weights -------------------------------------------------
  k_pack_w<<<ew_blocks(8LL * 3 * 32), B256, 0, stream>>>(pnW, 74, 128, 3, pnWB);
  k_pack_w<<<ew_blocks(8LL * 3 * 32), B256, 0, stream>>>(pe1W, 86, 128, 3, pe1WB);
  k_pack_w<<<ew_blocks(8LL * 4 * 32), B256, 0, stream>>>(etW, 128, 128, 4, etWB);
  k_pack_w<<<ew_blocks(24LL * 4 * 32), B256, 0, stream>>>(g0ihW, 128, 384, 4, g0ihWB);
  k_pack_w<<<ew_blocks(24LL * 4 * 32), B256, 0, stream>>>(g0hhW, 128, 384, 4, g0hhWB);

  // 1) hv = lrelu(nf @ Wpn + b)
  k_pack_a<<<ew_blocks(((N + 15LL) / 16) * 3 * 32), B256, 0, stream>>>(nf, 74, N, 74, 3, nfA);
  k_gemm_packed<<<gemm_blocks(N, 128), B256, 0, stream>>>(nfA, N, pnWB, 3, pnB, 128, hv, 1);
  // 2) he1 = lrelu(concat(nf[src], ef) @ Wpe1 + b)
  k_pack_a_edge<<<ew_blocks(((E + 15LL) / 16) * 3 * 32), B256, 0, stream>>>(nf, ef, src, E, 3, xeA);
  k_gemm_packed<<<gemm_blocks(E, 128), B256, 0, stream>>>(xeA, E, pe1WB, 3, pe1B, 128, he1, 1);
  // 3) logits = lrelu(concat(hv[dst], he1) @ Wpe2 + b)
  k_edge_logits<<<ew_blocks(E), B256, 0, stream>>>(hv, he1, 0, src, dst, pe2W, pe2B, logit, E);
  // 4) edge softmax over dst
  k_fill_u<<<ew_blocks(N), B256, 0, stream>>>(menc, NEG_INF_ENC, N);
  k_seg_max<<<ew_blocks(E), B256, 0, stream>>>(logit, dst, menc, E);
  k_fill_f<<<ew_blocks(N), B256, 0, stream>>>(ssum, 0.0f, N);
  k_exp_sum<<<ew_blocks(E), B256, 0, stream>>>(logit, dst, menc, attn, ssum, E);
  k_norm<<<ew_blocks(E), B256, 0, stream>>>(attn, ssum, dst, E);
  // 5) ctx = segsum(attn * (he1 @ Wet + b), dst)   (packed WMMA + scatter)
  k_pack_a<<<ew_blocks(((E + 15LL) / 16) * 4 * 32), B256, 0, stream>>>(he1, 128, E, 128, 4, he1A);
  k_fill_f<<<ew_blocks(NF), B256, 0, stream>>>(ctx, 0.0f, NF);
  k_gemm_msg_scatter<<<gemm_blocks(E, 128), B256, 0, stream>>>(he1A, E, etWB, 4, etB, attn, dst, ctx);
  // 6) ctx = elu(ctx)
  k_elu<<<ew_blocks(NF), B256, 0, stream>>>(ctx, NF);
  // 7) GRU0: h = relu(gru(ctx, hv))
  k_pack_a<<<ew_blocks(((N + 15LL) / 16) * 4 * 32), B256, 0, stream>>>(ctx, 128, N, 128, 4, ctxA);
  k_pack_a<<<ew_blocks(((N + 15LL) / 16) * 4 * 32), B256, 0, stream>>>(hv, 128, N, 128, 4, hvA);
  k_gemm_packed<<<gemm_blocks(N, 384), B256, 0, stream>>>(ctxA, N, g0ihWB, 4, g0ihB, 384, gi, 0);
  k_gemm_packed<<<gemm_blocks(N, 384), B256, 0, stream>>>(hvA, N, g0hhWB, 4, g0hhB, 384, gh, 0);
  k_gru<<<ew_blocks(NF), B256, 0, stream>>>(gi, gh, hv, h, NF);
  k_pack_a<<<ew_blocks(((N + 15LL) / 16) * 4 * 32), B256, 0, stream>>>(h, 128, N, 128, 4, hA);

  // layers 1..2
  for (int L = 0; L < 2; ++L) {
    int base = 17 + 8 * L;
    const float* peW    = (const float*)d_in[base + 0];
    const float* peB    = (const float*)d_in[base + 1];
    const float* pnodeW = (const float*)d_in[base + 2];
    const float* pnodeB = (const float*)d_in[base + 3];
    const float* gihW   = (const float*)d_in[base + 4];
    const float* gihB   = (const float*)d_in[base + 5];
    const float* ghhW   = (const float*)d_in[base + 6];
    const float* ghhB   = (const float*)d_in[base + 7];

    k_edge_logits<<<ew_blocks(E), B256, 0, stream>>>(h, h, 1, src, dst, peW, peB, logit, E);
    k_fill_u<<<ew_blocks(N), B256, 0, stream>>>(menc, NEG_INF_ENC, N);
    k_seg_max<<<ew_blocks(E), B256, 0, stream>>>(logit, dst, menc, E);
    k_fill_f<<<ew_blocks(N), B256, 0, stream>>>(ssum, 0.0f, N);
    k_exp_sum<<<ew_blocks(E), B256, 0, stream>>>(logit, dst, menc, attn, ssum, E);
    k_norm<<<ew_blocks(E), B256, 0, stream>>>(attn, ssum, dst, E);

    k_pack_w<<<ew_blocks(8LL * 4 * 32), B256, 0, stream>>>(pnodeW, 128, 128, 4, pnodeWB);
    k_gemm_packed<<<gemm_blocks(N, 128), B256, 0, stream>>>(hA, N, pnodeWB, 4, pnodeB, 128, hp, 0);
    k_fill_f<<<ew_blocks(NF), B256, 0, stream>>>(ctx, 0.0f, NF);
    k_scatter_ctx<<<ew_blocks(EF), B256, 0, stream>>>(hp, attn, src, dst, ctx, EF);
    k_elu<<<ew_blocks(NF), B256, 0, stream>>>(ctx, NF);

    k_pack_w<<<ew_blocks(24LL * 4 * 32), B256, 0, stream>>>(gihW, 128, 384, 4, lihWB);
    k_pack_w<<<ew_blocks(24LL * 4 * 32), B256, 0, stream>>>(ghhW, 128, 384, 4, lhhWB);
    k_pack_a<<<ew_blocks(((N + 15LL) / 16) * 4 * 32), B256, 0, stream>>>(ctx, 128, N, 128, 4, ctxA);
    k_gemm_packed<<<gemm_blocks(N, 384), B256, 0, stream>>>(ctxA, N, lihWB, 4, gihB, 384, gi, 0);
    k_gemm_packed<<<gemm_blocks(N, 384), B256, 0, stream>>>(hA, N, lhhWB, 4, ghhB, 384, gh, 0);
    k_gru<<<ew_blocks(NF), B256, 0, stream>>>(gi, gh, h, h, NF);
    k_pack_a<<<ew_blocks(((N + 15LL) / 16) * 4 * 32), B256, 0, stream>>>(h, 128, N, 128, 4, hA);
  }

  // graph mean-pool
  k_fill_f<<<ew_blocks((long long)G * 128), B256, 0, stream>>>(out, 0.0f, (long long)G * 128);
  k_fill_f<<<ew_blocks(G), B256, 0, stream>>>(counts, 0.0f, G);
  k_pool_add<<<ew_blocks(NF), B256, 0, stream>>>(h, gid, out, NF);
  k_counts<<<ew_blocks(N), B256, 0, stream>>>(gid, counts, N);
  k_finalize<<<ew_blocks((long long)G * 128), B256, 0, stream>>>(out, counts, (long long)G * 128);
}
